// GraphAttentionLayer_601295422005
// MI455X (gfx1250) — matrix operand order repaired and checked by reading, not compiled, not run on hardware
//
#include <hip/hip_runtime.h>

// GAT layer for MI455X (gfx1250, wave32, WMMA).
//   B=8, N=2048, HIDDEN=128, HEADS=4, HEAD_DIM=32
// Roofline: adj = 134 MB (~6us @ 23.3 TB/s); 134M exp transcendentals;
// 8.6 GFLOP aggregation -> f16 WMMA (co-executes with exp/VALU stream).
// Single fused pass: num/den softmax (no max shift needed; e bounded, self-loop
// guarantees den>0), so adj is read exactly once and each exp computed once.

#define GAT_B      8
#define GAT_N      2048
#define GAT_H      4
#define GAT_D      32
#define GAT_HID    128

typedef __attribute__((ext_vector_type(16))) _Float16 v16h;
typedef __attribute__((ext_vector_type(8)))  float    v8f;

union Frag16 { v16h h; uint4 u[2]; };
union FragC  { v8f  f; float e[8]; };

// ---------------------------------------------------------------------------
// Kernel 1: Wh = X @ W via WMMA; writes WhT_f16[b][c][j] (col-major for
// kernel-2 B fragments) + el[b][h][j], er[b][h][j] in fp32.
// Grid: B*(N/16) blocks, 256 threads (8 waves, one 16-col N-tile each).
// ---------------------------------------------------------------------------
__global__ __launch_bounds__(256) void gat_wh_kernel(
    const float* __restrict__ X, const float* __restrict__ W,
    const float* __restrict__ a,
    _Float16* __restrict__ WhT, float* __restrict__ el, float* __restrict__ er)
{
    // padded rows: 144 f16 = 288 B (16B-aligned b128 reads, reduced conflicts)
    __shared__ __align__(16) _Float16 wlds[GAT_HID][144]; // W^T as f16: [c][k]
    __shared__ __align__(16) _Float16 xlds[16][144];      // X tile f16: [j][k]
    __shared__ float ctile[GAT_HID][17];                  // C transpose: [c][j]

    const int t   = threadIdx.x;
    const int blk = blockIdx.x;
    const int b   = blk >> 7;          // /128
    const int j0  = (blk & 127) << 4;  // *16

    // Stage W (128x128 f32, k-major) -> LDS transposed f16 [c][k]
    for (int it = 0; it < 64; ++it) {
        int idx = t + it * 256;              // idx = k*128 + c
        int k = idx >> 7, c = idx & 127;
        wlds[c][k] = (_Float16)W[idx];
    }
    // Stage X tile (16 x 128 f32) -> LDS f16 [j][k]
    for (int it = 0; it < 8; ++it) {
        int idx = t + it * 256;              // idx = j*128 + k
        int i = idx >> 7, k = idx & 127;
        xlds[i][k] = (_Float16)X[((size_t)b * GAT_N + j0 + i) * GAT_HID + k];
    }
    __syncthreads();

    const int w    = t >> 5;       // wave id = N-tile (16 cols)
    const int lane = t & 31;
    const int hi   = lane >> 4;    // lane half selects K sub-block
    const int m    = lane & 15;
    const int cc   = (w << 4) + m; // this lane's output column

    v8f acc = {};
    #pragma unroll
    for (int kk = 0; kk < 4; ++kk) {         // K = 128 = 4 x 32
        Frag16 af, bf;
        // A (16x32 f16): lane half 0 -> K 0..15 ; half 1 -> K 16..31
        const _Float16* ap = &xlds[m][kk * 32 + hi * 16];
        af.u[0] = *(const uint4*)ap;
        af.u[1] = *(const uint4*)(ap + 8);
        // B (32x16 f16): lane half 0 -> K 0..15 ; half 1 -> K 16..31 (col = cc)
        const _Float16* bp = &wlds[cc][kk * 32 + hi * 16];
        bf.u[0] = *(const uint4*)bp;
        bf.u[1] = *(const uint4*)(bp + 8);
        acc = __builtin_amdgcn_wmma_f32_16x16x32_f16(
                  false, af.h, false, bf.h, (short)0, acc, false, false);
    }

    // C: VGPR r, lane -> (j_local = r + 8*hi, col = cc). Transpose via LDS.
    FragC cf; cf.f = acc;
    #pragma unroll
    for (int r = 0; r < 8; ++r)
        ctile[cc][r + hi * 8] = cf.e[r];
    __syncthreads();

    // Coalesced WhT_f16 write: thread pair per column row, 16B each.
    {
        int c = t >> 1, half = t & 1;
        union { _Float16 hv[8]; uint4 u; } o;
        #pragma unroll
        for (int q = 0; q < 8; ++q)
            o.hv[q] = (_Float16)ctile[c][half * 8 + q];
        *(uint4*)&WhT[((size_t)(b * GAT_HID + c)) * GAT_N + j0 + half * 8] = o.u;
    }

    // el/er from fp32 C tile (full precision into the exp path).
    if (t < 64) {
        int h = t >> 4, j = t & 15;
        float sl = 0.f, sr = 0.f;
        #pragma unroll
        for (int d = 0; d < GAT_D; ++d) {
            float v = ctile[h * GAT_D + d][j];
            sl += v * a[d];
            sr += v * a[GAT_D + d];
        }
        el[(b * GAT_H + h) * GAT_N + j0 + j] = sl;
        er[(b * GAT_H + h) * GAT_N + j0 + j] = sr;
    }
}

// ---------------------------------------------------------------------------
// Kernel 2: fused masked-softmax attention aggregation.
// Grid: B*(N/16) blocks (16 query rows each), 256 threads = 8 waves.
// Producer role (all 8 waves): rows 2w..2w+1, all 4 heads: p = adj ? exp(lrelu)
//   into double-buffered LDS f16 tile (WMMA-A layout friendly, row-major).
// Consumer role (wave w): head h=w/2, col-half nh=w&1: one v_wmma per 32-j tile.
// ---------------------------------------------------------------------------
__global__ __launch_bounds__(256) void gat_attn_kernel(
    const int* __restrict__ adj, const _Float16* __restrict__ WhT,
    const float* __restrict__ el, const float* __restrict__ er,
    float* __restrict__ out)
{
    // rows padded to 40 f16 = 80 B (16B-aligned, conflict-reduced b128 reads)
    __shared__ __align__(16) _Float16 pbuf[2][GAT_H][16][40];
    __shared__ float den[16][GAT_H];

    const int t    = threadIdx.x;
    const int lane = t & 31;
    const int w    = t >> 5;
    const int blk  = blockIdx.x;
    const int b    = blk >> 7;
    const int i0   = (blk & 127) << 4;

    const int h  = w >> 1, nh = w & 1;   // consumer mapping
    const int hi = lane >> 4, m = lane & 15;

    // el for this wave's 2 producer rows, all heads (uniform across lanes)
    float elv[2][GAT_H];
    #pragma unroll
    for (int r = 0; r < 2; ++r)
        #pragma unroll
        for (int hh = 0; hh < GAT_H; ++hh)
            elv[r][hh] = el[(b * GAT_H + hh) * GAT_N + i0 + w * 2 + r];

    float dsum[2][GAT_H] = {};

    const int*      adjb = adj + (size_t)b * GAT_N * GAT_N;
    const _Float16* whb  = WhT + (size_t)b * GAT_HID * GAT_N;
    const int ccol = h * GAT_D + nh * 16 + m;
    const _Float16* bptr = whb + (size_t)ccol * GAT_N + hi * 16; // + 32*jt

    auto produce = [&](int jt, int buf) {
        const int j = jt * 32 + lane;
        float erv[GAT_H];
        #pragma unroll
        for (int hh = 0; hh < GAT_H; ++hh)
            erv[hh] = er[(b * GAT_H + hh) * GAT_N + j];
        #pragma unroll
        for (int r = 0; r < 2; ++r) {
            const int il = w * 2 + r;
            const int av = adjb[(size_t)(i0 + il) * GAT_N + j]; // adj read once
            #pragma unroll
            for (int hh = 0; hh < GAT_H; ++hh) {
                float x = elv[r][hh] + erv[hh];
                x = x > 0.f ? x : 0.2f * x;              // leaky_relu(0.2)
                float p = (av > 0) ? __expf(x) : 0.f;    // one exp per element
                dsum[r][hh] += p;
                pbuf[buf][hh][il][lane] = (_Float16)p;
            }
        }
    };

    v8f acc = {};
    produce(0, 0);
    __syncthreads();

    for (int jt = 0; jt < 64; ++jt) {
        const int buf = jt & 1;
        if (jt + 2 < 64)  // prefetch next-next adj tile -> global_prefetch_b8
            __builtin_prefetch(&adjb[(size_t)(i0 + w * 2) * GAT_N + (jt + 2) * 32 + lane], 0, 1);
        if (jt + 1 < 64)
            produce(jt + 1, buf ^ 1);

        // A (16x32 f16) from LDS: row M=m, K chunks per lane half
        Frag16 af, bf;
        const _Float16* ap = &pbuf[buf][h][m][hi * 16];
        af.u[0] = *(const uint4*)ap;
        af.u[1] = *(const uint4*)(ap + 8);
        // B (32x16 f16) from WhT (col-major): 32 contiguous bytes per lane
        const _Float16* bp = bptr + jt * 32;
        bf.u[0] = *(const uint4*)bp;
        bf.u[1] = *(const uint4*)(bp + 8);
        acc = __builtin_amdgcn_wmma_f32_16x16x32_f16(
                  false, af.h, false, bf.h, (short)0, acc, false, false);

        __syncthreads();  // tile jt consumed everywhere; buf is safe to refill
    }

    // Reduce den across the 32 lanes of each wave, publish to LDS.
    #pragma unroll
    for (int r = 0; r < 2; ++r)
        #pragma unroll
        for (int hh = 0; hh < GAT_H; ++hh) {
            float v = dsum[r][hh];
            #pragma unroll
            for (int off = 16; off > 0; off >>= 1)
                v += __shfl_xor(v, off, 32);
            if (lane == 0) den[w * 2 + r][hh] = v;
        }
    __syncthreads();

    // out[b, i0+M, h*32 + nh*16 + m] = C[r]/den ; C: M = r + 8*hi, N = m
    FragC cf; cf.f = acc;
    #pragma unroll
    for (int r = 0; r < 8; ++r) {
        int M = r + hi * 8;
        float o = cf.e[r] / den[M][h];
        out[((size_t)(b * GAT_N + i0 + M)) * (GAT_H * GAT_D)
            + h * GAT_D + nh * 16 + m] = o;
    }
}

// ---------------------------------------------------------------------------
extern "C" void kernel_launch(void* const* d_in, const int* in_sizes, int n_in,
                              void* d_out, int out_size, void* d_ws, size_t ws_size,
                              hipStream_t stream) {
    (void)in_sizes; (void)n_in; (void)out_size; (void)ws_size;
    const float* X   = (const float*)d_in[0];
    const int*   adj = (const int*)  d_in[1];
    const float* W   = (const float*)d_in[2];
    const float* a   = (const float*)d_in[3];
    float*       out = (float*)d_out;

    char* ws = (char*)d_ws;
    _Float16* WhT = (_Float16*)ws;                                  // 4 MB
    float*    el  = (float*)(ws + (size_t)GAT_B * GAT_HID * GAT_N * sizeof(_Float16));
    float*    er  = el + (size_t)GAT_B * GAT_H * GAT_N;             // 256 KB each

    dim3 grid(GAT_B * (GAT_N / 16));   // 1024 blocks
    gat_wh_kernel  <<<grid, 256, 0, stream>>>(X, W, a, WhT, el, er);
    gat_attn_kernel<<<grid, 256, 0, stream>>>(adj, WhT, el, er, out);
}
